// RelationalHypergraphLayer_90744069030530
// MI455X (gfx1250) — matrix operand order repaired and checked by reading, not compiled, not run on hardware
//
#include <hip/hip_runtime.h>
#include <hip/hip_bf16.h>

#define N_NODES 100000
#define N_EDGES 1600000
#define IN_DIM  128
#define HEADS   8
#define D_HEAD  16
#define OUT_DIM 128
#define NEG_SLOPE 0.2f

typedef __attribute__((ext_vector_type(16))) __bf16 v16bf;
typedef __attribute__((ext_vector_type(8)))  float  v8f;

// ---------------------------------------------------------------------------
// Kernel 1: feat = h @ W via v_wmma_f32_16x16x32_bf16 (f32 accumulate).
// Block = 256 threads = 8 waves; block tile = 16 rows x 128 cols.
// Wave w computes the 16x16 tile for head w (cols [16w, 16w+16)).
// ---------------------------------------------------------------------------
__global__ __launch_bounds__(256) void gat_proj_wmma(
    const float* __restrict__ h, const float* __restrict__ W,
    float* __restrict__ feat)
{
  __shared__ float  sA[16][IN_DIM];           // 8 KB  : h tile, f32
  __shared__ __bf16 sWT[OUT_DIM][IN_DIM + 4]; // ~33 KB: W transposed, bf16

  const int tid  = threadIdx.x;
  const int row0 = blockIdx.x * 16;

  // Stage W (k-major in memory: W[k*128 + c]) transposed into LDS as bf16.
  for (int idx = tid; idx < IN_DIM * OUT_DIM; idx += 256) {
    int k = idx >> 7, c = idx & 127;
    sWT[c][k] = (__bf16)W[idx];
  }
  // Stage the 16x128 tile of h.
  for (int idx = tid; idx < 16 * IN_DIM; idx += 256) {
    int r = idx >> 7, c = idx & 127;
    sA[r][c] = h[(size_t)(row0 + r) * IN_DIM + c];
  }
  __syncthreads();

  const int wave = tid >> 5;   // 0..7 == head == 16-column tile
  const int lane = tid & 31;
  const int m    = lane & 15;  // A row / B column / C column
  const int kh   = lane >> 4;  // lane-half selector for K layout

  v8f acc = {};
  #pragma unroll
  for (int ks = 0; ks < 4; ++ks) {       // K = 128 in 4 steps of 32
    const int kb = ks * 32;
    v16bf a, b;
    // A 16x32 bf16 layout: lane half 0 -> K {kb..kb+7, kb+16..kb+23},
    //                      lane half 1 -> K {kb+8..kb+15, kb+24..kb+31}
    const float* pa0 = &sA[m][kb + kh * 8];
    const float* pa1 = &sA[m][kb + 16 + kh * 8];
    #pragma unroll
    for (int j = 0; j < 8; ++j) {
      a[j]     = (__bf16)pa0[j];
      a[8 + j] = (__bf16)pa1[j];
    }
    // B 32x16 bf16 layout: lane = column (m), same K split as A.
    const __bf16* pb0 = &sWT[wave * 16 + m][kb + kh * 8];
    const __bf16* pb1 = &sWT[wave * 16 + m][kb + 16 + kh * 8];
    #pragma unroll
    for (int i = 0; i < 8; ++i) { b[i] = pb0[i]; b[8 + i] = pb1[i]; }

    acc = __builtin_amdgcn_wmma_f32_16x16x32_bf16(
        false, a, false, b, (short)0, acc, false, false);
  }

  // C/D layout: element r = D[row0 + r + 8*kh][wave*16 + m]
  #pragma unroll
  for (int r = 0; r < 8; ++r) {
    int row = row0 + r + 8 * kh;
    feat[(size_t)row * OUT_DIM + wave * 16 + m] = acc[r];
  }
}

// ---------------------------------------------------------------------------
// Ordered-uint encoding of float so unsigned atomicMax == float max.
// ---------------------------------------------------------------------------
__device__ __forceinline__ unsigned f2ord(float f) {
  unsigned u = __float_as_uint(f);
  return (u & 0x80000000u) ? ~u : (u | 0x80000000u);
}
__device__ __forceinline__ float ord2f(unsigned u) {
  return __uint_as_float((u & 0x80000000u) ? (u & 0x7fffffffu) : ~u);
}

// ---------------------------------------------------------------------------
// Kernel 2: per (node, head): el/er dots, init emax/esum, out = h + bias.
// ---------------------------------------------------------------------------
__global__ __launch_bounds__(256) void gat_node_init(
    const float* __restrict__ feat, const float* __restrict__ h,
    const float* __restrict__ attn_l, const float* __restrict__ attn_r,
    const float* __restrict__ bias,
    float* __restrict__ el, float* __restrict__ er,
    unsigned* __restrict__ emax, float* __restrict__ esum,
    float* __restrict__ out)
{
  int t = blockIdx.x * 256 + threadIdx.x;
  if (t >= N_NODES * HEADS) return;
  int n = t >> 3, hd = t & 7;

  const float* f  = feat   + (size_t)n * OUT_DIM + hd * D_HEAD;
  const float* al = attn_l + hd * D_HEAD;
  const float* ar = attn_r + hd * D_HEAD;
  float sl = 0.f, sr = 0.f;
  #pragma unroll
  for (int d = 0; d < D_HEAD; ++d) { float v = f[d]; sl += v * al[d]; sr += v * ar[d]; }
  el[t] = sl;
  er[t] = sr;
  emax[t] = 0u;    // ordered encoding below every finite float
  esum[t] = 0.f;

  const float* hp = h    + (size_t)n * IN_DIM + hd * D_HEAD;
  const float* bp = bias + hd * D_HEAD;
  float*       op = out  + (size_t)n * OUT_DIM + hd * D_HEAD;
  #pragma unroll
  for (int d = 0; d < D_HEAD; ++d) op[d] = hp[d] + bp[d];
}

// ---------------------------------------------------------------------------
// Kernel 3: per (edge, head): segment max of leaky-relu score into dst.
// ---------------------------------------------------------------------------
__global__ __launch_bounds__(256) void gat_edge_max(
    const int* __restrict__ src, const int* __restrict__ dst,
    const float* __restrict__ el, const float* __restrict__ er,
    unsigned* __restrict__ emax)
{
  int t = blockIdx.x * 256 + threadIdx.x;
  if (t >= N_EDGES * HEADS) return;
  int e = t >> 3, hd = t & 7;
  int s = src[e], d = dst[e];
  float v = el[s * HEADS + hd] + er[d * HEADS + hd];
  v = v > 0.f ? v : NEG_SLOPE * v;
  atomicMax(&emax[d * HEADS + hd], f2ord(v));
}

// ---------------------------------------------------------------------------
// Kernel 4: per (edge, head): exp(score - max), store, segment-sum to dst.
// ---------------------------------------------------------------------------
__global__ __launch_bounds__(256) void gat_edge_exp(
    const int* __restrict__ src, const int* __restrict__ dst,
    const float* __restrict__ el, const float* __restrict__ er,
    const unsigned* __restrict__ emax,
    float* __restrict__ aexp, float* __restrict__ esum)
{
  int t = blockIdx.x * 256 + threadIdx.x;
  if (t >= N_EDGES * HEADS) return;
  int e = t >> 3, hd = t & 7;
  int s = src[e], d = dst[e];
  float v = el[s * HEADS + hd] + er[d * HEADS + hd];
  v = v > 0.f ? v : NEG_SLOPE * v;
  float mx = ord2f(emax[d * HEADS + hd]);
  float ex = __expf(v - mx);
  aexp[t] = ex;
  atomicAdd(&esum[d * HEADS + hd], ex);
}

// ---------------------------------------------------------------------------
// Kernel 5: per (edge, 4-channel group): normalize, weight feat[src],
// scatter-add into out[dst]. 32 threads cover one edge's 128 channels.
// feat (51 MB) is L2-resident on MI455X (192 MB L2) -> gathers hit L2.
// ---------------------------------------------------------------------------
__global__ __launch_bounds__(256) void gat_scatter(
    const int* __restrict__ src, const int* __restrict__ dst,
    const float* __restrict__ feat, const float* __restrict__ aexp,
    const float* __restrict__ esum, float* __restrict__ out)
{
  int t = blockIdx.x * 256 + threadIdx.x;
  if (t >= N_EDGES * 32) return;
  int e = t >> 5, q = t & 31;       // q*4 = head*16 + (q&3)*4
  int hd = q >> 2;
  int s = src[e], d = dst[e];

  const float* fp = feat + (size_t)s * OUT_DIM + q * 4;
  __builtin_prefetch(fp, 0, 0);      // global_prefetch_b8

  float a = aexp[e * HEADS + hd] / esum[d * HEADS + hd];
  float4 f4 = *(const float4*)fp;
  float* op = out + (size_t)d * OUT_DIM + q * 4;
  atomicAdd(op + 0, a * f4.x);
  atomicAdd(op + 1, a * f4.y);
  atomicAdd(op + 2, a * f4.z);
  atomicAdd(op + 3, a * f4.w);
}

// ---------------------------------------------------------------------------
extern "C" void kernel_launch(void* const* d_in, const int* in_sizes, int n_in,
                              void* d_out, int out_size, void* d_ws, size_t ws_size,
                              hipStream_t stream) {
  const float* h      = (const float*)d_in[0];
  const float* W      = (const float*)d_in[1];
  const float* attn_l = (const float*)d_in[2];
  const float* attn_r = (const float*)d_in[3];
  const float* bias   = (const float*)d_in[4];
  const int*   src    = (const int*)d_in[5];
  const int*   dst    = (const int*)d_in[6];
  float*       out    = (float*)d_out;

  // Workspace layout (~115 MB):
  float*    feat = (float*)d_ws;                               // N*128
  float*    el   = feat + (size_t)N_NODES * OUT_DIM;           // N*8
  float*    er   = el   + (size_t)N_NODES * HEADS;             // N*8
  unsigned* emax = (unsigned*)(er + (size_t)N_NODES * HEADS);  // N*8
  float*    esum = (float*)(emax + (size_t)N_NODES * HEADS);   // N*8
  float*    aexp = esum + (size_t)N_NODES * HEADS;             // E*8

  gat_proj_wmma<<<N_NODES / 16, 256, 0, stream>>>(h, W, feat);
  gat_node_init<<<(N_NODES * HEADS + 255) / 256, 256, 0, stream>>>(
      feat, h, attn_l, attn_r, bias, el, er, emax, esum, out);
  gat_edge_max<<<(N_EDGES * HEADS + 255) / 256, 256, 0, stream>>>(
      src, dst, el, er, emax);
  gat_edge_exp<<<(N_EDGES * HEADS + 255) / 256, 256, 0, stream>>>(
      src, dst, el, er, emax, aexp, esum);
  gat_scatter<<<(N_EDGES * 32 + 255) / 256, 256, 0, stream>>>(
      src, dst, feat, aexp, esum, out);
}
